// MaskedMultiHeadAttention_41558103556114
// MI455X (gfx1250) — matrix operand order repaired and checked by reading, not compiled, not run on hardware
//
#include <hip/hip_runtime.h>
#include <hip/hip_bf16.h>

// ---------------- problem constants (from reference) ----------------
#define D_MODEL 1024
#define NUM_HEADS 16
#define D_HEAD 64
#define BATCH 2
#define SEQ 2048
#define M_TOTAL (BATCH * SEQ)   // 4096 rows in flattened [B*S, D] activations

typedef __attribute__((ext_vector_type(16))) __bf16 v16bf;
typedef __attribute__((ext_vector_type(8)))  float  v8f;
typedef __attribute__((ext_vector_type(8)))  unsigned short us8;

// ---------------- bf16 helpers (round-to-nearest-even) ----------------
__device__ __forceinline__ unsigned short f32_to_bf16(float f) {
  union { float f; unsigned int u; } x; x.f = f;
  unsigned int r = x.u + 0x7FFFu + ((x.u >> 16) & 1u);
  return (unsigned short)(r >> 16);
}
__device__ __forceinline__ float bf16_to_f32(unsigned short h) {
  union { unsigned int u; float f; } x; x.u = ((unsigned int)h) << 16;
  return x.f;
}

union FragU { v16bf v; us8 h[2]; unsigned short u[16]; };

// A-matrix fragment (16x32 bf16), ISA 7.12.2 layout:
//   lane L: row M = L&15, half = L>>4
//   elements 0..7  -> K = k0 +      half*8 + {0..7}
//   elements 8..15 -> K = k0 + 16 + half*8 + {0..7}
// `row` must already point at this lane's matrix row (global or LDS).
__device__ __forceinline__ v16bf load_frag_a(const unsigned short* row, int k0, int lane) {
  const int half = (lane >> 4) & 1;
  FragU f;
  f.h[0] = *(const us8*)(row + k0 + half * 8);
  f.h[1] = *(const us8*)(row + k0 + 16 + half * 8);
  return f.v;
}

// B-matrix fragment (32x16 bf16): lane L holds column N = L&15, half = L>>4;
//   elements 0..7 -> K = k0 + half*8 + {0..7}, elements 8..15 -> K = k0 + 16 + half*8 + {0..7}.
// `col` must point at the K-contiguous storage of this lane's column
// (for y = x @ W^T that is simply row N of W).
__device__ __forceinline__ v16bf load_frag_b(const unsigned short* col, int k0, int lane) {
  const int half = (lane >> 4) & 1;
  FragU f;
  f.h[0] = *(const us8*)(col + k0 + half * 8);
  f.h[1] = *(const us8*)(col + k0 + 16 + half * 8);
  return f.v;
}

__device__ __forceinline__ v8f wmma_bf16(v16bf a, v16bf b, v8f c) {
  return __builtin_amdgcn_wmma_f32_16x16x32_bf16(
      /*neg_a=*/false, a, /*neg_b=*/false, b,
      /*c_mod=*/(short)0, c, /*reuse_a=*/false, /*reuse_b=*/false);
}

// Async global->LDS copy of one 16-byte chunk per lane (ASYNCcnt-tracked DMA path).
__device__ __forceinline__ void async_copy_b128(const void* gsrc, const void* lds_dst) {
  // generic LDS address: low 32 bits == LDS byte offset (ISA 10.2 aperture rule)
  unsigned loff = (unsigned)(unsigned long long)lds_dst;
  asm volatile("global_load_async_to_lds_b128 %0, %1, off"
               :: "v"(loff), "v"(gsrc) : "memory");
}
__device__ __forceinline__ void wait_async0() {
  asm volatile("s_wait_asynccnt 0" ::: "memory");
}

// ---------------- f32 -> bf16 convert (vectorized x4) ----------------
__global__ __launch_bounds__(256) void convert_bf16_kernel(
    const float* __restrict__ src, unsigned short* __restrict__ dst, int n4) {
  int i = blockIdx.x * blockDim.x + threadIdx.x;   // one float4 per thread
  if (i < n4) {
    const float4 v = ((const float4*)src)[i];
    ushort4 o;
    o.x = f32_to_bf16(v.x); o.y = f32_to_bf16(v.y);
    o.z = f32_to_bf16(v.z); o.w = f32_to_bf16(v.w);
    ((ushort4*)dst)[i] = o;
  }
}

// ---------------- RoPE on [B*H, S, 64] bf16 tensor ----------------
__global__ __launch_bounds__(256) void rope_kernel(unsigned short* __restrict__ T) {
  int t = blockIdx.x * blockDim.x + threadIdx.x;     // B*H*S*32 threads
  if (t >= BATCH * NUM_HEADS * SEQ * 32) return;
  int i  = t & 31;                  // rotary pair index (0..31)
  int s  = (t >> 5) % SEQ;          // sequence position
  int bh = t / (32 * SEQ);
  unsigned short* p = T + ((size_t)bh * SEQ + s) * D_HEAD;
  float fr  = __powf(10000.0f, -((float)(2 * i)) * (1.0f / 64.0f));  // inv_freq
  float ang = (float)s * fr;
  float c  = __cosf(ang);
  float sn = __sinf(ang);
  float u = bf16_to_f32(p[i]);
  float v = bf16_to_f32(p[i + 32]);
  p[i]      = f32_to_bf16(u * c - v * sn);
  p[i + 32] = f32_to_bf16(v * c + u * sn);
}

// ---------------- WMMA GEMM: Y = X @ W^T + b ----------------
// Block = 128 threads (4 waves) owns a 64x64 output tile: wave w computes rows
// w*16..w*16+15 across all 64 columns. Per k-step of 64, 64x64 X and W tiles are
// staged into double-buffered LDS with global_load_async_to_lds_b128 (next
// buffer issued before computing on the current one). All 10 fragments of a
// k-step are loaded up front so the 8 WMMAs issue back-to-back.
// LDS rows padded to 72 elements (36 banks, coprime pattern) -> conflict-free
// 16-lane ds_load_b128, still 16B-aligned for the async b128 destinations.
// mode 0: store bf16 to [B, H, S, Dh]       (Q, K)
// mode 1: store bf16 to [B, H, Dh, S]       (V transposed, for PV B-fragments)
// mode 2: store f32  to [M_TOTAL, 1024]     (final output projection)
#define LSTR 72   // padded LDS row stride in elements (144 B)
__global__ __launch_bounds__(128) void gemm_bf16_kernel(
    const unsigned short* __restrict__ X,
    const unsigned short* __restrict__ W,
    const float* __restrict__ bias,
    unsigned short* __restrict__ out_bf,
    float* __restrict__ out_f32,
    int mode) {
  __shared__ __align__(16) unsigned short Xs[2][64 * LSTR];   // 2 x 9 KB
  __shared__ __align__(16) unsigned short Ws[2][64 * LSTR];   // 2 x 9 KB

  const int tid  = threadIdx.x;
  const int lane = tid & 31;
  const int wave = tid >> 5;
  const int half = (lane >> 4) & 1;
  const int lr   = lane & 15;

  const int NBN = D_MODEL / 64;                 // 16 column blocks
  const int m0  = (blockIdx.x / NBN) * 64;
  const int n0  = (blockIdx.x % NBN) * 64;

  // stage one 64x64 X tile + 64x64 W tile into LDS buffer `buf` (async)
  auto stage = [&](int buf, int k0) {
    // 64 rows x 128 bytes = 512 16-byte chunks per tile; 4 per thread per tile
#pragma unroll
    for (int c = tid; c < 512; c += 128) {
      const int row = c >> 3;            // 0..63
      const int q   = (c & 7) * 8;       // element offset within row (8 el = 16B)
      async_copy_b128(X + (size_t)(m0 + row) * D_MODEL + k0 + q,
                      &Xs[buf][row * LSTR + q]);
      async_copy_b128(W + (size_t)(n0 + row) * D_MODEL + k0 + q,
                      &Ws[buf][row * LSTR + q]);
    }
  };

  v8f acc0 = {}, acc1 = {}, acc2 = {}, acc3 = {};   // wave's 16x64 f32 tile

  stage(0, 0);
  wait_async0();
  __syncthreads();

  for (int k0 = 0; k0 < D_MODEL; k0 += 64) {
    const int cur = (k0 >> 6) & 1;
    if (k0 + 64 < D_MODEL) stage(cur ^ 1, k0 + 64);   // prefetch next buffer

    // hoisted fragment loads: 20 ds_load_b128 in flight, then 8 WMMAs
    const unsigned short* xrow = &Xs[cur][(wave * 16 + lr) * LSTR];
    const v16bf a0  = load_frag_a(xrow, 0, lane);
    const v16bf a1  = load_frag_a(xrow, 32, lane);
    const unsigned short* w0 = &Ws[cur][(0 * 16 + lr) * LSTR];
    const unsigned short* w1 = &Ws[cur][(1 * 16 + lr) * LSTR];
    const unsigned short* w2 = &Ws[cur][(2 * 16 + lr) * LSTR];
    const unsigned short* w3 = &Ws[cur][(3 * 16 + lr) * LSTR];
    const v16bf b00 = load_frag_b(w0, 0, lane), b01 = load_frag_b(w0, 32, lane);
    const v16bf b10 = load_frag_b(w1, 0, lane), b11 = load_frag_b(w1, 32, lane);
    const v16bf b20 = load_frag_b(w2, 0, lane), b21 = load_frag_b(w2, 32, lane);
    const v16bf b30 = load_frag_b(w3, 0, lane), b31 = load_frag_b(w3, 32, lane);

    acc0 = wmma_bf16(a0, b00, acc0);
    acc1 = wmma_bf16(a0, b10, acc1);
    acc2 = wmma_bf16(a0, b20, acc2);
    acc3 = wmma_bf16(a0, b30, acc3);
    acc0 = wmma_bf16(a1, b01, acc0);
    acc1 = wmma_bf16(a1, b11, acc1);
    acc2 = wmma_bf16(a1, b21, acc2);
    acc3 = wmma_bf16(a1, b31, acc3);

    if (k0 + 64 < D_MODEL) wait_async0();   // next buffer landed (this wave)
    __syncthreads();                        // all waves' copies visible
  }

  // ---- epilogue: C/D layout -> memory (VGPR j holds row half*8+j, col = lane) ----
  const int mrow = m0 + wave * 16 + half * 8;
#pragma unroll
  for (int t = 0; t < 4; ++t) {
    const v8f acc = (t == 0) ? acc0 : (t == 1) ? acc1 : (t == 2) ? acc2 : acc3;
    const int e = n0 + t * 16 + lr;
    const float bv = bias[e];
#pragma unroll
    for (int j = 0; j < 8; ++j) {
      const int m = mrow + j;
      float v = acc[j] + bv;
      if (mode == 2) {
        out_f32[(size_t)m * D_MODEL + e] = v;
      } else {
        const int b = m / SEQ, s = m % SEQ;
        const int h = e / D_HEAD, d = e % D_HEAD;
        if (mode == 0)
          out_bf[(((size_t)(b * NUM_HEADS + h)) * SEQ + s) * D_HEAD + d] = f32_to_bf16(v);
        else
          out_bf[(((size_t)(b * NUM_HEADS + h)) * D_HEAD + d) * SEQ + s] = f32_to_bf16(v);
      }
    }
  }
}

// ---------------- Flash attention (causal) ----------------
// Qb, Kb: [B,H,S,64] bf16; Vt: [B,H,64,S] bf16 (transposed); ctx: [B*S, 1024] bf16.
// One wave owns one 16-row query tile; loops key blocks of 32. K and V fragment
// loads are hoisted to the top of the block so V is in flight during softmax.
#define PSTR 40   // padded P-tile row stride (80 B = 20 banks, conflict-free)
__global__ __launch_bounds__(128) void attn_kernel(
    const unsigned short* __restrict__ Qb,
    const unsigned short* __restrict__ Kb,
    const unsigned short* __restrict__ Vt,
    unsigned short* __restrict__ ctx) {
  __shared__ __align__(16) unsigned short lds[4][16 * PSTR];  // P tile per wave

  const int lane = threadIdx.x & 31;
  const int wave = threadIdx.x >> 5;
  const int w    = blockIdx.x * 4 + wave;        // 4096 waves total
  const int QT   = SEQ / 16;                     // 128 query tiles per (b,h)
  const int qt   = w % QT;
  const int bh   = w / QT;                       // b*H + h
  const int h    = bh % NUM_HEADS;
  const int b    = bh / NUM_HEADS;
  const int q0   = qt * 16;
  const int half = (lane >> 4) & 1;
  const int lr   = lane & 15;

  const unsigned short* Qh = Qb + ((size_t)bh * SEQ) * D_HEAD;
  const unsigned short* Kh = Kb + ((size_t)bh * SEQ) * D_HEAD;
  const unsigned short* Vh = Vt + ((size_t)bh * D_HEAD) * SEQ;
  unsigned short* ldsw = &lds[wave][0];

  // Q fragments for this tile (Dh = 64 -> two K-chunks of 32)
  const unsigned short* qrow = Qh + (size_t)(q0 + lr) * D_HEAD;
  const v16bf qa0 = load_frag_a(qrow, 0, lane);
  const v16bf qa1 = load_frag_a(qrow, 32, lane);

  v8f o0 = {}, o1 = {}, o2 = {}, o3 = {};       // 16x64 f32 accumulator
  float rowmax[8], rowsum[8];
#pragma unroll
  for (int j = 0; j < 8; ++j) { rowmax[j] = -__builtin_inff(); rowsum[j] = 0.f; }

  const float LOG2E = 1.4426950408889634f;
  const float SCALE = 0.125f;                   // 1/sqrt(64)

  for (int kb = 0; kb < q0 + 16; kb += 32) {    // causal: only blocks with k <= q0+15
    if (kb + 32 < q0 + 16) {                    // hint next key block into cache
      __builtin_prefetch(Kh + (size_t)(kb + 32 + lr) * D_HEAD, 0, 1);
      __builtin_prefetch(Vh + (size_t)lr * SEQ + kb + 32, 0, 1);
    }
    // ---- hoisted global fragment loads: K (scores) and V (PV) ----
    const unsigned short* kcol0 = Kh + (size_t)(kb + lr) * D_HEAD;
    const unsigned short* kcol1 = Kh + (size_t)(kb + 16 + lr) * D_HEAD;
    const v16bf kb00 = load_frag_b(kcol0, 0, lane), kb01 = load_frag_b(kcol0, 32, lane);
    const v16bf kb10 = load_frag_b(kcol1, 0, lane), kb11 = load_frag_b(kcol1, 32, lane);
    const v16bf vb0 = load_frag_b(Vh + (size_t)(0 * 16 + lr) * SEQ + kb, 0, lane);
    const v16bf vb1 = load_frag_b(Vh + (size_t)(1 * 16 + lr) * SEQ + kb, 0, lane);
    const v16bf vb2 = load_frag_b(Vh + (size_t)(2 * 16 + lr) * SEQ + kb, 0, lane);
    const v16bf vb3 = load_frag_b(Vh + (size_t)(3 * 16 + lr) * SEQ + kb, 0, lane);

    // ---- scores: two 16-key sub-tiles, Dh reduced in two WMMAs each ----
    v8f c0 = {}, c1 = {};
    c0 = wmma_bf16(qa0, kb00, c0);
    c1 = wmma_bf16(qa0, kb10, c1);
    c0 = wmma_bf16(qa1, kb01, c0);
    c1 = wmma_bf16(qa1, kb11, c1);

    // ---- online softmax in the C/D register layout ----
#pragma unroll
    for (int j = 0; j < 8; ++j) {
      const int row = q0 + half * 8 + j;
      const int k0i = kb + lr;
      const int k1i = kb + 16 + lr;
      float s0 = (k0i <= row) ? c0[j] * SCALE : -__builtin_inff();
      float s1 = (k1i <= row) ? c1[j] * SCALE : -__builtin_inff();
      float mx = fmaxf(s0, s1);
#pragma unroll
      for (int off = 1; off < 16; off <<= 1) mx = fmaxf(mx, __shfl_xor(mx, off, 32));
      const float nm    = fmaxf(rowmax[j], mx);
      const float alpha = exp2f((rowmax[j] - nm) * LOG2E);
      rowmax[j] = nm;
      const float p0 = exp2f((s0 - nm) * LOG2E);
      const float p1 = exp2f((s1 - nm) * LOG2E);
      float ps = p0 + p1;
#pragma unroll
      for (int off = 1; off < 16; off <<= 1) ps += __shfl_xor(ps, off, 32);
      rowsum[j] = rowsum[j] * alpha + ps;
      o0[j] *= alpha; o1[j] *= alpha; o2[j] *= alpha; o3[j] *= alpha;
      // stage P (bf16) row-major 16x32 (padded) in LDS for A-fragment re-read
      ldsw[(half * 8 + j) * PSTR + lr]      = f32_to_bf16(p0);
      ldsw[(half * 8 + j) * PSTR + 16 + lr] = f32_to_bf16(p1);
    }

    // LDS is in-order per wave; drain DS writes before cross-lane re-read
    asm volatile("s_wait_dscnt 0" ::: "memory");
    const v16bf pa = load_frag_a(ldsw + (size_t)lr * PSTR, 0, lane);

    // ---- accumulate P @ V (V transposed -> contiguous B-fragments) ----
    o0 = wmma_bf16(pa, vb0, o0);
    o1 = wmma_bf16(pa, vb1, o1);
    o2 = wmma_bf16(pa, vb2, o2);
    o3 = wmma_bf16(pa, vb3, o3);
  }

  // ---- normalize and store context [B*S, 1024] bf16 ----
#pragma unroll
  for (int j = 0; j < 8; ++j) {
    const float inv = 1.0f / rowsum[j];
    const size_t m  = (size_t)b * SEQ + q0 + half * 8 + j;
    unsigned short* dst = ctx + m * D_MODEL + h * D_HEAD;
    dst[0 * 16 + lr] = f32_to_bf16(o0[j] * inv);
    dst[1 * 16 + lr] = f32_to_bf16(o1[j] * inv);
    dst[2 * 16 + lr] = f32_to_bf16(o2[j] * inv);
    dst[3 * 16 + lr] = f32_to_bf16(o3[j] * inv);
  }
}

// ---------------- host side ----------------
extern "C" void kernel_launch(void* const* d_in, const int* in_sizes, int n_in,
                              void* d_out, int out_size, void* d_ws, size_t ws_size,
                              hipStream_t stream) {
  (void)in_sizes; (void)n_in; (void)out_size; (void)ws_size;
  const float* x  = (const float*)d_in[0];
  const float* Wq = (const float*)d_in[1];
  const float* bq = (const float*)d_in[2];
  const float* Wk = (const float*)d_in[3];
  const float* bk = (const float*)d_in[4];
  const float* Wv = (const float*)d_in[5];
  const float* bv = (const float*)d_in[6];
  const float* Wo = (const float*)d_in[7];
  const float* bo = (const float*)d_in[8];
  float* out = (float*)d_out;

  // workspace layout (bf16 = 2 bytes)
  const size_t MB = 1024 * 1024;
  char* ws = (char*)d_ws;
  unsigned short* xb   = (unsigned short*)(ws + 0);        //  8 MB  [4096,1024]
  unsigned short* wqb  = (unsigned short*)(ws + 8  * MB);  //  2 MB
  unsigned short* wkb  = (unsigned short*)(ws + 10 * MB);  //  2 MB
  unsigned short* wvb  = (unsigned short*)(ws + 12 * MB);  //  2 MB
  unsigned short* wob  = (unsigned short*)(ws + 14 * MB);  //  2 MB
  unsigned short* Qb   = (unsigned short*)(ws + 16 * MB);  //  8 MB  [B,H,S,64]
  unsigned short* Kb   = (unsigned short*)(ws + 24 * MB);  //  8 MB  [B,H,S,64]
  unsigned short* Vtb  = (unsigned short*)(ws + 32 * MB);  //  8 MB  [B,H,64,S]
  unsigned short* ctxb = (unsigned short*)(ws + 40 * MB);  //  8 MB  [4096,1024]

  const int n_x = M_TOTAL * D_MODEL;        // 4,194,304
  const int n_w = D_MODEL * D_MODEL;        // 1,048,576

  // 1) convert inputs to bf16 (float4-vectorized)
  convert_bf16_kernel<<<n_x / 4 / 256, 256, 0, stream>>>(x,  xb,  n_x / 4);
  convert_bf16_kernel<<<n_w / 4 / 256, 256, 0, stream>>>(Wq, wqb, n_w / 4);
  convert_bf16_kernel<<<n_w / 4 / 256, 256, 0, stream>>>(Wk, wkb, n_w / 4);
  convert_bf16_kernel<<<n_w / 4 / 256, 256, 0, stream>>>(Wv, wvb, n_w / 4);
  convert_bf16_kernel<<<n_w / 4 / 256, 256, 0, stream>>>(Wo, wob, n_w / 4);

  // 2) QKV projections: 64x64 tiles per block, 4 waves/block
  const int gemm_blocks = (M_TOTAL / 64) * (D_MODEL / 64);   // 1024
  gemm_bf16_kernel<<<gemm_blocks, 128, 0, stream>>>(xb, wqb, bq, Qb,  nullptr, 0);
  gemm_bf16_kernel<<<gemm_blocks, 128, 0, stream>>>(xb, wkb, bk, Kb,  nullptr, 0);
  gemm_bf16_kernel<<<gemm_blocks, 128, 0, stream>>>(xb, wvb, bv, Vtb, nullptr, 1);

  // 3) RoPE on Q and K
  const int n_rope = BATCH * NUM_HEADS * SEQ * 32;           // 2,097,152
  rope_kernel<<<n_rope / 256, 256, 0, stream>>>(Qb);
  rope_kernel<<<n_rope / 256, 256, 0, stream>>>(Kb);

  // 4) causal flash attention (4096 waves, 4 waves/block)
  attn_kernel<<<BATCH * NUM_HEADS * (SEQ / 16) / 4, 128, 0, stream>>>(Qb, Kb, Vtb, ctxb);

  // 5) output projection straight into f32 d_out
  gemm_bf16_kernel<<<gemm_blocks, 128, 0, stream>>>(ctxb, wob, bo, nullptr, out, 2);
}
